// WaveletHeterogeneousAttention_54417235640351
// MI455X (gfx1250) — matrix operand order repaired and checked by reading, not compiled, not run on hardware
//
#include <hip/hip_runtime.h>
#include <hip/hip_bf16.h>
#include <math.h>

typedef _Float16 f16;
typedef __attribute__((ext_vector_type(16))) _Float16 v16h;
typedef __attribute__((ext_vector_type(8)))  _Float16 v8h;
typedef __attribute__((ext_vector_type(8)))  float    v8f;

#define B_  2
#define C_  128
#define H_  256
#define W_  256
#define NH_ 8
#define HD_ 16
#define HH_ 128
#define WH_ 128
#define LH_ (HH_*WH_)    // 16384
#define HW_ (H_*W_)      // 65536

static constexpr size_t S_half = (size_t)B_*C_*HH_*WH_;   // 4,194,304
static constexpr size_t S_full = (size_t)B_*C_*H_*W_;     // 16,777,216

// ---------------- async-LDS probe (gfx1250 ASYNCcnt path) ----------------
#if __has_builtin(__builtin_amdgcn_global_load_async_to_lds_b128) && \
    __has_builtin(__builtin_amdgcn_global_store_async_from_lds_b128)
#define HAVE_ASYNC 1
#else
#define HAVE_ASYNC 0
#endif

#if HAVE_ASYNC
// diagnostic-confirmed param type: pointer to 'int __attribute__((vector_size(16)))'
typedef int v4i __attribute__((vector_size(16)));
typedef __attribute__((address_space(1))) v4i* gasync_p;   // global side
typedef __attribute__((address_space(3))) v4i* lasync_p;   // LDS side
#if __has_builtin(__builtin_amdgcn_s_wait_asynccnt)
#define WAIT_ASYNC() __builtin_amdgcn_s_wait_asynccnt(0)
#else
#define WAIT_ASYNC() asm volatile("s_wait_asynccnt 0x0" ::: "memory")
#endif
#endif

// ---------------- workspace layout (offsets in floats) ----------------
static constexpr size_t OFF_OUTB = 0;                    // q first, later attn-output
static constexpr size_t OFF_V    = OFF_OUTB + S_full;
static constexpr size_t OFF_KS   = OFF_V + S_full;       // 4 x S_half
static constexpr size_t OFF_LL   = OFF_KS + 4*S_half;
static constexpr size_t OFF_LH   = OFF_LL + S_half;
static constexpr size_t OFF_HL   = OFF_LH + S_half;
static constexpr size_t OFF_HH   = OFF_HL + S_half;
static constexpr size_t OFF_T0   = OFF_HH + S_half;
static constexpr size_t OFF_QLH  = OFF_T0 + S_half;
static constexpr size_t OFF_QHL  = OFF_QLH + S_half;
static constexpr size_t OFF_QHH  = OFF_QHL + S_half;
static constexpr size_t OFF_CAT  = OFF_QHH + S_half;     // (B, 256, 128, 128): rain ch0-127, haze ch128-255
static constexpr size_t OFF_QLL  = OFF_CAT + 2*S_half;
static constexpr size_t OFF_NRM  = OFF_QLL + S_half;     // 512 floats used (reciprocal norms)
static constexpr size_t OFF_ATTG = OFF_NRM + 1024;
static constexpr size_t OFF_ATTLH= OFF_ATTG + 4096;
static constexpr size_t OFF_ATTHL= OFF_ATTLH + 4096;
static constexpr size_t OFF_ATTHH= OFF_ATTHL + 4096;
static constexpr size_t OFF_ATTM = OFF_ATTHH + 4096;
static constexpr size_t OFF_F16  = OFF_ATTM + 4096;      // f16 weight region starts here

// f16 weight sub-offsets (in halfs), layout [k=ky*KW+kx][co][ci]
static constexpr size_t HO_GLR = 0;
static constexpr size_t HO_GLZ = HO_GLR + (size_t)4*9*128*128;
static constexpr size_t HO_GLH = HO_GLZ + (size_t)4*9*128*128;
static constexpr size_t HO_GHL = HO_GLH + (size_t)2*15*128*128;
static constexpr size_t HO_GHH = HO_GHL + (size_t)2*15*128*128;
static constexpr size_t HO_PW1 = HO_GHH + (size_t)2*9*128*128;
static constexpr size_t HO_CAT = HO_PW1 + (size_t)128*128;
static constexpr size_t HO_PROJ= HO_CAT + (size_t)128*256;

// ---------------- depthwise 3x3 stride1 pad1 ----------------
__global__ __launch_bounds__(256) void dw3x3_s1(const float* __restrict__ in,
                                                const float* __restrict__ w,
                                                const float* __restrict__ bias,
                                                float* __restrict__ out) {
  size_t idx = (size_t)blockIdx.x*256 + threadIdx.x;
  if (idx >= S_full) return;
  int x = (int)(idx & 255), y = (int)((idx>>8)&255);
  int ch = (int)((idx>>16)&127), b = (int)(idx>>23);
  const float* wp = w + ch*9;
  const float* ip = in + (((size_t)(b*C_+ch))<<16);
  float s = bias[ch];
  #pragma unroll
  for (int ky=0; ky<3; ky++) {
    int yy = y + ky - 1; if ((unsigned)yy >= 256u) continue;
    #pragma unroll
    for (int kx=0; kx<3; kx++) {
      int xx = x + kx - 1; if ((unsigned)xx >= 256u) continue;
      s += wp[ky*3+kx] * ip[yy*256 + xx];
    }
  }
  out[idx] = s;
}

// ---------------- depthwise 3x3 stride2 pad1 ----------------
__global__ __launch_bounds__(256) void dw3x3_s2(const float* __restrict__ in,
                                                const float* __restrict__ w,
                                                const float* __restrict__ bias,
                                                float* __restrict__ out) {
  size_t idx = (size_t)blockIdx.x*256 + threadIdx.x;
  if (idx >= S_half) return;
  int x = (int)(idx & 127), y = (int)((idx>>7)&127);
  int ch = (int)((idx>>14)&127), b = (int)(idx>>21);
  const float* wp = w + ch*9;
  const float* ip = in + (((size_t)(b*C_+ch))<<16);
  float s = bias[ch];
  #pragma unroll
  for (int ky=0; ky<3; ky++) {
    int yy = 2*y + ky - 1; if ((unsigned)yy >= 256u) continue;
    #pragma unroll
    for (int kx=0; kx<3; kx++) {
      int xx = 2*x + kx - 1; if ((unsigned)xx >= 256u) continue;
      s += wp[ky*3+kx] * ip[yy*256 + xx];
    }
  }
  out[idx] = s;
}

// ---------------- Haar DWT ----------------
__global__ __launch_bounds__(256) void dwt_haar_k(const float* __restrict__ q,
                                                  float* __restrict__ ll, float* __restrict__ lh,
                                                  float* __restrict__ hl, float* __restrict__ hh) {
  size_t idx = (size_t)blockIdx.x*256 + threadIdx.x;
  if (idx >= S_half) return;
  int x = (int)(idx & 127), y = (int)((idx>>7)&127);
  size_t bc = idx >> 14;                        // b*C + ch
  const float* ip = q + (bc<<16);
  float a = ip[(2*y)*256 + 2*x];
  float b2= ip[(2*y)*256 + 2*x+1];
  float c = ip[(2*y+1)*256 + 2*x];
  float d = ip[(2*y+1)*256 + 2*x+1];
  ll[idx] = (a+b2+c+d)*0.5f;
  lh[idx] = (c+d-a-b2)*0.5f;
  hl[idx] = (b2+d-a-c)*0.5f;
  hh[idx] = (a+d-b2-c)*0.5f;
}

// ---------------- weight transform OIHW f32 -> [k][co][ci] f16 ----------------
__global__ __launch_bounds__(256) void wt_k(const float* __restrict__ w, f16* __restrict__ out,
                                            int CO, int CI, int KH, int KW) {
  int i = blockIdx.x*256 + threadIdx.x;
  int total = CO*CI*KH*KW;
  if (i >= total) return;
  int kx = i % KW; int ky = (i/KW) % KH;
  int ci = (i/(KW*KH)) % CI; int co = i/(KW*KH*CI);
  out[((size_t)((ky*KW+kx)*CO + co))*CI + ci] = (f16)w[i];
}

// ---------------- WMMA implicit-GEMM conv ----------------
// Tile: M = MT*16 pixels (one row), N = 128 output channels (8 waves x 16 co).
// B-fragment (weights) reused across MT accumulators; kx/cb loops fully unrolled
// so fragment loads pipeline underneath the WMMA stream.
template<int KH, int KW, int PH, int PW, int CIN, int MT, bool GATED>
__global__ __launch_bounds__(256) void conv_wmma_k(
    const float* __restrict__ in, int ICT, int in_off, int Hc, int Wc,
    const f16* __restrict__ wA, const float* __restrict__ bA,
    const f16* __restrict__ wB, const float* __restrict__ bB,
    float* __restrict__ outp, int OCT, int out_off,
    const float* __restrict__ ep_mask, const float* __restrict__ ep_resid) {
  constexpr int MPIX = MT*16;
  constexpr int PIXT = MPIX + 2*PW;
  constexpr int STR  = CIN + 8;                 // pad (272B rows) keeps b128 alignment, spreads banks
  __shared__ f16 lds[KH*PIXT*STR];

  const int tid = threadIdx.x;
  const int Wt = Wc / MPIX;
  int bidx = blockIdx.x;
  const int xt = bidx % Wt; bidx /= Wt;
  const int y  = bidx % Hc; const int b = bidx / Hc;
  const int x0 = xt * MPIX;

  // stage the KH x (MPIX+2PW) x CIN input patch to LDS as f16
  for (int e = tid; e < KH*PIXT*CIN; e += 256) {
    int p  = e % PIXT;
    int t2 = e / PIXT;
    int ci = t2 % CIN;
    int r  = t2 / CIN;
    int yy = y + r - PH;
    int xx = x0 + p - PW;
    float v = 0.f;
    if ((unsigned)yy < (unsigned)Hc && (unsigned)xx < (unsigned)Wc)
      v = in[((size_t)(b*ICT + in_off + ci)*Hc + yy)*Wc + xx];
    lds[(r*PIXT + p)*STR + ci] = (f16)v;
  }
  __syncthreads();

  const int wv   = tid >> 5;
  const int lane = tid & 31;
  const int n    = lane & 15;                   // A: M row (pixel); B: N col (co)
  const int hi   = lane >> 4;
  const int hi8  = hi * 8;
  const int co   = (wv << 4) + n;

  v8f acc1[MT], acc2[MT];
  #pragma unroll
  for (int t = 0; t < MT; t++) {
    acc1[t] = (v8f){0.f,0.f,0.f,0.f,0.f,0.f,0.f,0.f};
    acc2[t] = (v8f){0.f,0.f,0.f,0.f,0.f,0.f,0.f,0.f};
  }

  for (int r = 0; r < KH; ++r) {
    #pragma unroll
    for (int kx = 0; kx < KW; ++kx) {
      const int kk = r*KW + kx;
      const f16* wrowA = wA + ((size_t)(kk*128 + co))*CIN;
      const f16* wrowB = GATED ? (wB + ((size_t)(kk*128 + co))*CIN) : (const f16*)0;
      if (kk + 1 < KH*KW) {                     // prefetch next tap's weight row (global_prefetch_b8)
        __builtin_prefetch(wrowA + (size_t)128*CIN, 0, 1);
        if (GATED) __builtin_prefetch(wrowB + (size_t)128*CIN, 0, 1);
      }
      #pragma unroll
      for (int cb = 0; cb < CIN; cb += 32) {
        // A fragments: lane<16 -> K {cb..cb+7, cb+16..cb+23}; lane>=16 -> +8
        v16h a[MT];
        #pragma unroll
        for (int t = 0; t < MT; t++) {
          const f16* arow = &lds[(r*PIXT + n + t*16 + kx)*STR];
          v8h a0 = *(const v8h*)(arow + cb + hi8);
          v8h a1 = *(const v8h*)(arow + cb + 16 + hi8);
          #pragma unroll
          for (int j=0;j<8;j++){ a[t][j]=a0[j]; a[t][j+8]=a1[j]; }
        }
        // B fragment: K {cb+hi8..+7, cb+16+hi8..+7} (matches A)
        {
          v8h w0 = *(const v8h*)(wrowA + cb + hi8);
          v8h w1 = *(const v8h*)(wrowA + cb + 16 + hi8);
          v16h bm;
          #pragma unroll
          for (int j=0;j<8;j++){ bm[j]=w0[j]; bm[j+8]=w1[j]; }
          #pragma unroll
          for (int t = 0; t < MT; t++)
            acc1[t] = __builtin_amdgcn_wmma_f32_16x16x32_f16(false, a[t], false, bm, (short)0, acc1[t], false, false);
        }
        if (GATED) {
          v8h g0 = *(const v8h*)(wrowB + cb + hi8);
          v8h g1 = *(const v8h*)(wrowB + cb + 16 + hi8);
          v16h gm;
          #pragma unroll
          for (int j=0;j<8;j++){ gm[j]=g0[j]; gm[j+8]=g1[j]; }
          #pragma unroll
          for (int t = 0; t < MT; t++)
            acc2[t] = __builtin_amdgcn_wmma_f32_16x16x32_f16(false, a[t], false, gm, (short)0, acc2[t], false, false);
        }
      }
    }
  }

  const float biasA = bA[co];
  const float biasB = GATED ? bB[co] : 0.f;
  #pragma unroll
  for (int t = 0; t < MT; t++) {
    #pragma unroll
    for (int r2 = 0; r2 < 8; ++r2) {
      int m  = r2 + hi8;                        // D: M = r + 8*hi, N = lane&15
      int xx = x0 + t*16 + m;
      float v = acc1[t][r2] + biasA;
      if (GATED) {
        float g = acc2[t][r2] + biasB;
        v = v * (1.f / (1.f + expf(-g)));
      }
      size_t oidx = ((size_t)(b*OCT + out_off + co)*Hc + y)*Wc + xx;
      if (ep_mask)  v *= ep_mask[((size_t)(b*128 + co)*Hc + y)*Wc + xx];
      if (ep_resid) v += ep_resid[oidx];
      outp[oidx] = v;
    }
  }
}

// ---------------- per-row reciprocal norms ----------------
__global__ __launch_bounds__(256) void norms_k(const float* __restrict__ qll,
                                               const float* __restrict__ k0,
                                               float* __restrict__ nrm) {
  __shared__ float red[256];
  int row = blockIdx.x;                         // 0..255 = q rows, 256..511 = k rows
  const float* src = (row < 256) ? (qll + (size_t)row*LH_) : (k0 + (size_t)(row-256)*LH_);
  float s = 0.f;
  for (int i = threadIdx.x; i < LH_; i += 256) { float v = src[i]; s += v*v; }
  red[threadIdx.x] = s; __syncthreads();
  for (int o = 128; o > 0; o >>= 1) {
    if (threadIdx.x < o) red[threadIdx.x] += red[threadIdx.x+o];
    __syncthreads();
  }
  if (threadIdx.x == 0) nrm[row] = 1.f / fmaxf(sqrtf(red[0]), 1e-12f);
}

// ---------------- global channel attention: WMMA Gram (K=16384 split over 8 waves) + softmax ----------------
__global__ __launch_bounds__(256) void gram_softmax_k(const float* __restrict__ qll,
                                                      const float* __restrict__ k0,
                                                      const float* __restrict__ nrm,
                                                      const float* __restrict__ scale_p,
                                                      float* __restrict__ attn) {
  __shared__ float part[8*256];
  __shared__ float sm[256];
  const int bh = blockIdx.x;
  const int b = bh / NH_, h = bh % NH_;
  const int tid = threadIdx.x;
  const int wv = tid >> 5, lane = tid & 31, n = lane & 15, hi = lane >> 4, hi8 = hi*8;
  const float* qrow = qll + ((size_t)(b*C_ + h*HD_ + n))*LH_;
  const float* krow = k0  + ((size_t)(b*C_ + h*HD_ + n))*LH_;
  v8f acc = {0.f,0.f,0.f,0.f,0.f,0.f,0.f,0.f};
  const int cb0 = wv * 2048;
  #pragma unroll 4
  for (int cb = cb0; cb < cb0 + 2048; cb += 32) {
    v16h a, bm;
    #pragma unroll
    for (int j=0;j<8;j++){ a[j]=(f16)qrow[cb+hi8+j]; a[j+8]=(f16)qrow[cb+16+hi8+j]; }
    #pragma unroll
    for (int j=0;j<8;j++){ bm[j]=(f16)krow[cb+hi8+j]; bm[j+8]=(f16)krow[cb+16+hi8+j]; }
    acc = __builtin_amdgcn_wmma_f32_16x16x32_f16(false, a, false, bm, (short)0, acc, false, false);
  }
  #pragma unroll
  for (int r=0;r<8;r++) part[wv*256 + (r+hi8)*16 + n] = acc[r];
  __syncthreads();
  {
    float s = 0.f;
    #pragma unroll
    for (int w=0; w<8; w++) s += part[w*256 + tid];
    int c = tid >> 4, d = tid & 15;
    float rq = nrm[b*C_ + h*HD_ + c];
    float rk = nrm[256 + b*C_ + h*HD_ + d];
    sm[tid] = s * (*scale_p) * rq * rk;
  }
  __syncthreads();
  if (tid < 16) {
    int c = tid;
    float mx = -1e30f;
    for (int d=0; d<16; d++) mx = fmaxf(mx, sm[c*16+d]);
    float ssum = 0.f;
    for (int d=0; d<16; d++) ssum += expf(sm[c*16+d]-mx);
    float inv = 1.f / ssum;
    for (int d=0; d<16; d++) attn[(size_t)bh*256 + c*16 + d] = expf(sm[c*16+d]-mx)*inv;
  }
}

// ---------------- windowed channel attention (4x4 windows), softmax + mean via atomics ----------------
__global__ __launch_bounds__(256) void win_attn_k(const float* __restrict__ qv,
                                                  const float* __restrict__ kv,
                                                  const float* __restrict__ scale_p,
                                                  float* __restrict__ attn_acc) {
  __shared__ float qs[16][17], ks_[16][17];
  __shared__ float nq[16], nk[16], sm[256], rmx[16], rsm[16];
  const int b = blockIdx.x >> 10;
  const int win = blockIdx.x & 1023;
  const int wy = win >> 5, wx = win & 31;
  const int h = blockIdx.y;
  const int t = threadIdx.x;
  const int c = t >> 4, lp = t & 15;
  const int iy = wy*4 + (lp>>2), ix = wx*4 + (lp&3);
  size_t base = ((size_t)(b*C_ + h*HD_ + c)*HH_ + iy)*WH_ + ix;
  qs[c][lp]  = qv[base];
  ks_[c][lp] = kv[base];
  __syncthreads();
  if (t < 32) {
    int r = t & 15; float s = 0.f;
    if (t < 16) { for (int l=0;l<16;l++) s += qs[r][l]*qs[r][l];  nq[r] = 1.f/fmaxf(sqrtf(s), 1e-12f); }
    else        { for (int l=0;l<16;l++) s += ks_[r][l]*ks_[r][l]; nk[r] = 1.f/fmaxf(sqrtf(s), 1e-12f); }
  }
  __syncthreads();
  {
    int d = lp; float s = 0.f;
    for (int l=0;l<16;l++) s += qs[c][l]*ks_[d][l];
    sm[t] = s * nq[c] * nk[d] * (*scale_p);
  }
  __syncthreads();
  if (t < 16) {
    float mx = -1e30f;
    for (int d=0; d<16; d++) mx = fmaxf(mx, sm[t*16+d]);
    float ssum = 0.f;
    for (int d=0; d<16; d++) ssum += expf(sm[t*16+d]-mx);
    rmx[t] = mx; rsm[t] = 1.f/ssum;
  }
  __syncthreads();
  float val = expf(sm[t]-rmx[c]) * rsm[c] * (1.f/1024.f);
  atomicAdd(&attn_acc[((size_t)(b*NH_+h))*256 + t], val);
}

__global__ void zero_k(float* p, int n) {
  int i = blockIdx.x*256 + threadIdx.x;
  if (i < n) p[i] = 0.f;
}

// ---------------- mix attn heads + top-8 row sparsify ----------------
__global__ __launch_bounds__(256) void mix_sparsify_k(const float* __restrict__ g,
                                                      const float* __restrict__ alh,
                                                      const float* __restrict__ ahl,
                                                      const float* __restrict__ ahh,
                                                      const float* __restrict__ wmix,
                                                      float* __restrict__ out) {
  __shared__ float sm[256];
  __shared__ float thr[16];
  const int bh = blockIdx.x;
  const int t = threadIdx.x;
  size_t base = (size_t)bh*256 + t;
  float m = wmix[0]*g[base] + wmix[1]*alh[base] + wmix[2]*ahl[base] + wmix[3]*ahh[base];
  sm[t] = m;
  __syncthreads();
  if (t < 16) {
    float row[16]; bool used[16];
    for (int j=0;j<16;j++){ row[j]=sm[t*16+j]; used[j]=false; }
    float th = -1e30f;
    for (int pass=0; pass<8; pass++) {
      float mx = -1e30f; int mi = 0;
      for (int j=0;j<16;j++) if (!used[j] && row[j] > mx) { mx = row[j]; mi = j; }
      used[mi] = true; th = mx;
    }
    thr[t] = th;
  }
  __syncthreads();
  out[base] = (m >= thr[t>>4]) ? m : 0.f;
}

// ---------------- out = attn @ v + v*mask_gp ----------------
__global__ __launch_bounds__(256) void apply_attn_k(const float* __restrict__ attn,
                                                    const float* __restrict__ v,
                                                    const float* __restrict__ mask,
                                                    float* __restrict__ out) {
  __shared__ float am[256];
  const int bh = blockIdx.y;
  const int b = bh >> 3, h = bh & 7;
  am[threadIdx.x] = attn[(size_t)bh*256 + threadIdx.x];
  __syncthreads();
  size_t l = (size_t)blockIdx.x*256 + threadIdx.x;
  size_t rowbase = ((size_t)(b*C_ + h*HD_))*(size_t)HW_ + l;
  float vv[16];
  #pragma unroll
  for (int d=0; d<16; d++) vv[d] = v[rowbase + (size_t)d*HW_];
  #pragma unroll
  for (int c=0; c<16; c++) {
    float s = 0.f;
    #pragma unroll
    for (int d=0; d<16; d++) s += am[c*16+d]*vv[d];
    size_t oi = rowbase + (size_t)c*HW_;
    out[oi] = s + vv[c]*mask[oi];
  }
}

// ---------------- prior passthrough: async global->LDS->global when available ----------------
__global__ __launch_bounds__(256) void copy_prior_k(const float* __restrict__ src,
                                                    float* __restrict__ dst) {
#if HAVE_ASYNC
  __shared__ float buf[1024];
  const int t = threadIdx.x;
  size_t base = (size_t)blockIdx.x*1024 + (size_t)t*4;
  v4i* sg = (v4i*)(const_cast<float*>(src) + base);   // generic -> v4i*
  v4i* dg = (v4i*)(dst + base);
  v4i* sl = (v4i*)(buf + t*4);
  __builtin_amdgcn_global_load_async_to_lds_b128((gasync_p)sg, (lasync_p)sl, 0, 0);
  WAIT_ASYNC();
  __syncthreads();
  __builtin_amdgcn_global_store_async_from_lds_b128((gasync_p)dg, (lasync_p)sl, 0, 0);
  WAIT_ASYNC();
#else
  size_t base = (size_t)blockIdx.x*1024 + (size_t)threadIdx.x*4;
  float4 v = *(const float4*)(src + base);
  *(float4*)(dst + base) = v;
#endif
}

// ======================================================================
extern "C" void kernel_launch(void* const* d_in, const int* in_sizes, int n_in,
                              void* d_out, int out_size, void* d_ws, size_t ws_size,
                              hipStream_t stream) {
  const float* x        = (const float*)d_in[0];
  const float* prior    = (const float*)d_in[1];
  const float* mask_gp  = (const float*)d_in[2];
  const float* mask_ll  = (const float*)d_in[3];
  const float* mask_lh  = (const float*)d_in[4];
  const float* mask_hl  = (const float*)d_in[5];
  const float* qw       = (const float*)d_in[6];
  const float* qb       = (const float*)d_in[7];
  const float* kw       = (const float*)d_in[8];
  const float* kb       = (const float*)d_in[9];
  const float* vw       = (const float*)d_in[10];
  const float* vb       = (const float*)d_in[11];
  const float* glr_w    = (const float*)d_in[12];
  const float* glr_b    = (const float*)d_in[13];
  const float* glz_w    = (const float*)d_in[14];
  const float* glz_b    = (const float*)d_in[15];
  const float* glh_w    = (const float*)d_in[16];
  const float* glh_b    = (const float*)d_in[17];
  const float* ghl_w    = (const float*)d_in[18];
  const float* ghl_b    = (const float*)d_in[19];
  const float* ghh_w    = (const float*)d_in[20];
  const float* ghh_b    = (const float*)d_in[21];
  const float* pw1_w    = (const float*)d_in[22];
  const float* pw1_b    = (const float*)d_in[23];
  const float* cat_w    = (const float*)d_in[24];
  const float* cat_b    = (const float*)d_in[25];
  const float* scale    = (const float*)d_in[26];
  const float* wmix     = (const float*)d_in[27];
  const float* proj_w   = (const float*)d_in[28];
  const float* proj_b   = (const float*)d_in[29];
  (void)in_sizes; (void)n_in; (void)ws_size; (void)out_size;

  float* wsf  = (float*)d_ws;
  f16*   wsh  = (f16*)(wsf + OFF_F16);
  float* outf = (float*)d_out;

  float* q    = wsf + OFF_OUTB;   // becomes attn-output after DWT
  float* v    = wsf + OFF_V;
  float* ks0  = wsf + OFF_KS + 0*S_half;
  float* ks1  = wsf + OFF_KS + 1*S_half;
  float* ks2  = wsf + OFF_KS + 2*S_half;
  float* ks3  = wsf + OFF_KS + 3*S_half;
  float* ll   = wsf + OFF_LL;
  float* lh   = wsf + OFF_LH;
  float* hl   = wsf + OFF_HL;
  float* hh   = wsf + OFF_HH;
  float* t0   = wsf + OFF_T0;
  float* qlh  = wsf + OFF_QLH;
  float* qhl  = wsf + OFF_QHL;
  float* qhh  = wsf + OFF_QHH;
  float* cat  = wsf + OFF_CAT;
  float* qll  = wsf + OFF_QLL;
  float* nrm  = wsf + OFF_NRM;
  float* attg = wsf + OFF_ATTG;
  float* attlh= wsf + OFF_ATTLH;
  float* atthl= wsf + OFF_ATTHL;
  float* atthh= wsf + OFF_ATTHH;
  float* attm = wsf + OFF_ATTM;

  // 1) depthwise q, v
  dw3x3_s1<<<(unsigned)(S_full/256), 256, 0, stream>>>(x, qw, qb, q);
  dw3x3_s1<<<(unsigned)(S_full/256), 256, 0, stream>>>(x, vw, vb, v);
  // 2) depthwise stride-2 k's
  float* ksarr[4] = {ks0, ks1, ks2, ks3};
  for (int i = 0; i < 4; i++)
    dw3x3_s2<<<(unsigned)(S_half/256), 256, 0, stream>>>(x, kw + (size_t)i*C_*9, kb + i*C_, ksarr[i]);
  // 3) DWT
  dwt_haar_k<<<(unsigned)(S_half/256), 256, 0, stream>>>(q, ll, lh, hl, hh);

  // 4) weight transforms (OIHW f32 -> [k][co][ci] f16)
  auto wt = [&](const float* src, f16* dst, int CO, int CI, int KH, int KW) {
    int tot = CO*CI*KH*KW;
    wt_k<<<(tot+255)/256, 256, 0, stream>>>(src, dst, CO, CI, KH, KW);
  };
  for (int j = 0; j < 4; j++) wt(glr_w + (size_t)j*C_*C_*9,  wsh + HO_GLR + (size_t)j*9*128*128,  128, 128, 3, 3);
  for (int j = 0; j < 4; j++) wt(glz_w + (size_t)j*C_*C_*9,  wsh + HO_GLZ + (size_t)j*9*128*128,  128, 128, 3, 3);
  for (int j = 0; j < 2; j++) wt(glh_w + (size_t)j*C_*C_*15, wsh + HO_GLH + (size_t)j*15*128*128, 128, 128, 3, 5);
  for (int j = 0; j < 2; j++) wt(ghl_w + (size_t)j*C_*C_*15, wsh + HO_GHL + (size_t)j*15*128*128, 128, 128, 5, 3);
  for (int j = 0; j < 2; j++) wt(ghh_w + (size_t)j*C_*C_*9,  wsh + HO_GHH + (size_t)j*9*128*128,  128, 128, 3, 3);
  wt(pw1_w,  wsh + HO_PW1,  128, 128, 1, 1);
  wt(cat_w,  wsh + HO_CAT,  128, 256, 1, 1);
  wt(proj_w, wsh + HO_PROJ, 128, 128, 1, 1);

  // grid sizes: spatial convs use MT=2 (M=32), 1x1 convs use MT=4 (M=64)
  const unsigned gHalf2 = (unsigned)(B_ * HH_ * (WH_/32));   // 1024 tiles
  const unsigned gHalf4 = (unsigned)(B_ * HH_ * (WH_/64));   // 512 tiles
  const unsigned gFull4 = (unsigned)(B_ * H_  * (W_/64));    // 2048 tiles

  // 5) gated chains (WMMA)
  // glr: ll -> t0 -> cat[rain]
  conv_wmma_k<3,3,1,1,128,2,true><<<gHalf2,256,0,stream>>>(ll, 128, 0, HH_, WH_,
      wsh + HO_GLR + 0*147456, glr_b + 0,   wsh + HO_GLR + 1*147456, glr_b + 128,
      t0, 128, 0, nullptr, nullptr);
  conv_wmma_k<3,3,1,1,128,2,true><<<gHalf2,256,0,stream>>>(t0, 128, 0, HH_, WH_,
      wsh + HO_GLR + 2*147456, glr_b + 256, wsh + HO_GLR + 3*147456, glr_b + 384,
      cat, 256, 0, nullptr, nullptr);
  // glz: ll -> t0 -> cat[haze]
  conv_wmma_k<3,3,1,1,128,2,true><<<gHalf2,256,0,stream>>>(ll, 128, 0, HH_, WH_,
      wsh + HO_GLZ + 0*147456, glz_b + 0,   wsh + HO_GLZ + 1*147456, glz_b + 128,
      t0, 128, 0, nullptr, nullptr);
  conv_wmma_k<3,3,1,1,128,2,true><<<gHalf2,256,0,stream>>>(t0, 128, 0, HH_, WH_,
      wsh + HO_GLZ + 2*147456, glz_b + 256, wsh + HO_GLZ + 3*147456, glz_b + 384,
      cat, 256, 128, nullptr, nullptr);
  // glh (3x5 pad(1,2)) with mask_rp_lh fused
  conv_wmma_k<3,5,1,2,128,2,true><<<gHalf2,256,0,stream>>>(lh, 128, 0, HH_, WH_,
      wsh + HO_GLH + 0*245760, glh_b + 0, wsh + HO_GLH + 1*245760, glh_b + 128,
      qlh, 128, 0, mask_lh, nullptr);
  // ghl (5x3 pad(2,1)) with mask_rp_hl fused
  conv_wmma_k<5,3,2,1,128,2,true><<<gHalf2,256,0,stream>>>(hl, 128, 0, HH_, WH_,
      wsh + HO_GHL + 0*245760, ghl_b + 0, wsh + HO_GHL + 1*245760, ghl_b + 128,
      qhl, 128, 0, mask_hl, nullptr);
  // ghh (3x3)
  conv_wmma_k<3,3,1,1,128,2,true><<<gHalf2,256,0,stream>>>(hh, 128, 0, HH_, WH_,
      wsh + HO_GHH + 0*147456, ghh_b + 0, wsh + HO_GHH + 1*147456, ghh_b + 128,
      qhh, 128, 0, nullptr, nullptr);

  // 6) pw1 residual (in-place on cat halves): out = in + conv(in)*mask_ll
  conv_wmma_k<1,1,0,0,128,4,false><<<gHalf4,256,0,stream>>>(cat, 256, 0, HH_, WH_,
      wsh + HO_PW1, pw1_b, nullptr, nullptr, cat, 256, 0, mask_ll, cat);
  conv_wmma_k<1,1,0,0,128,4,false><<<gHalf4,256,0,stream>>>(cat, 256, 128, HH_, WH_,
      wsh + HO_PW1, pw1_b, nullptr, nullptr, cat, 256, 128, mask_ll, cat);

  // 7) cat conv (2C -> C)
  conv_wmma_k<1,1,0,0,256,4,false><<<gHalf4,256,0,stream>>>(cat, 256, 0, HH_, WH_,
      wsh + HO_CAT, cat_b, nullptr, nullptr, qll, 128, 0, nullptr, nullptr);

  // 8) global channel attention (WMMA Gram over L=16384)
  norms_k<<<512, 256, 0, stream>>>(qll, ks0, nrm);
  gram_softmax_k<<<B_*NH_, 256, 0, stream>>>(qll, ks0, nrm, scale, attg);

  // 9) windowed attention x3
  zero_k<<<(3*4096+255)/256, 256, 0, stream>>>(attlh, 3*4096);
  dim3 wg(B_*1024, NH_);
  win_attn_k<<<wg, 256, 0, stream>>>(qlh, ks1, scale, attlh);
  win_attn_k<<<wg, 256, 0, stream>>>(qhl, ks2, scale, atthl);
  win_attn_k<<<wg, 256, 0, stream>>>(qhh, ks3, scale, atthh);

  // 10) mix + sparsify
  mix_sparsify_k<<<B_*NH_, 256, 0, stream>>>(attg, attlh, atthl, atthh, wmix, attm);

  // 11) apply attention: out = attn @ v + v*mask_gp   (reuses the q buffer)
  dim3 ag(HW_/256, B_*NH_);
  apply_attn_k<<<ag, 256, 0, stream>>>(attm, v, mask_gp, q);

  // 12) final 1x1 projection (WMMA) straight to d_out
  conv_wmma_k<1,1,0,0,128,4,false><<<gFull4,256,0,stream>>>(q, 128, 0, H_, W_,
      wsh + HO_PROJ, proj_b, nullptr, nullptr, outf, 128, 0, nullptr, nullptr);

  // 13) prior passthrough (async-LDS path when toolchain exposes it)
  copy_prior_k<<<(unsigned)((B_*H_*W_)/1024), 256, 0, stream>>>(prior, outf + S_full);
}
